// ModGRU_45217415692492
// MI455X (gfx1250) — matrix operand order repaired and checked by reading, not compiled, not run on hardware
//
#include <hip/hip_runtime.h>
#include <hip/hip_bf16.h>

// ---------------------------------------------------------------------------
// Types for CDNA5 WMMA (wave32, 16x16x32 bf16 -> f32)
// ---------------------------------------------------------------------------
typedef __attribute__((ext_vector_type(8)))  float          v8f;
typedef __attribute__((ext_vector_type(16))) __bf16         v16bf;
typedef __attribute__((ext_vector_type(8)))  unsigned short v8us;
typedef __attribute__((ext_vector_type(16))) unsigned short v16us;

#define T_STEPS 512
#define BATCH   64
#define DIM     1024
#define HID     1024
#define H3      3072   // 3*HID

// round-to-nearest-even float -> bf16 bits
static __device__ __forceinline__ unsigned short f2bf(float f) {
    unsigned int u = __float_as_uint(f);
    unsigned int r = u + 0x7FFFu + ((u >> 16) & 1u);
    return (unsigned short)(r >> 16);
}

static __device__ __forceinline__ float sigmoidf_(float x) {
    return 1.0f / (1.0f + __expf(-x));
}

// ---------------------------------------------------------------------------
// Fragment loaders, contiguous k order (not the ISA table order); A and B agree
// A 16x32 (MxK), row-major bf16 source, leading dim = lda:
//   lane<16 : row=lane,    K = [0..15]
//   lane>=16: row=lane-16, K = [16..31]
// ---------------------------------------------------------------------------
static __device__ __forceinline__ v16bf load_a_frag(const unsigned short* __restrict__ A,
                                                    long lda, int mBase, int kBase, int lane) {
    int row  = mBase + (lane & 15);
    int koff = (lane >> 4) * 16;
    const unsigned short* p = A + (long)row * lda + kBase + koff;
    v8us lo = *(const v8us*)(p);
    v8us hi = *(const v8us*)(p + 8);
    v16us v = __builtin_shufflevector(lo, hi, 0,1,2,3,4,5,6,7,8,9,10,11,12,13,14,15);
    return __builtin_bit_cast(v16bf, v);
}

// B 32x16 (KxN), pre-packed: lane holds col n=lane%16, K=(lane/16)*16..+15,
// contiguous 16 ushorts at ((nTile*kTiles + kTile)*32 + lane)*16
static __device__ __forceinline__ v16bf load_b_frag(const unsigned short* __restrict__ Bp,
                                                    int kTiles, int nTile, int kTile, int lane) {
    const unsigned short* p = Bp + ((((long)nTile * kTiles + kTile) * 32) + lane) * 16;
    v16us v = *(const v16us*)p;
    return __builtin_bit_cast(v16bf, v);
}

#define WMMA_BF16(acc, a, b) \
    acc = __builtin_amdgcn_wmma_f32_16x16x32_bf16(false, a, false, b, (short)0, acc, false, false)

// ---------------------------------------------------------------------------
// Setup kernels
// ---------------------------------------------------------------------------

// Pack fp32 row-major W[K][N] -> bf16 B-fragment layout
__global__ void pack_w_kernel(const float* __restrict__ W, unsigned short* __restrict__ dst,
                              int K, int N) {
    long idx = (long)blockIdx.x * blockDim.x + threadIdx.x;
    if (idx >= (long)K * N) return;
    int k = (int)(idx / N);
    int n = (int)(idx % N);
    int kTiles = K >> 5;
    int nTile  = n >> 4;
    int kTile  = k >> 5;
    int lane   = (n & 15) + (((k & 31) >= 16) ? 16 : 0);
    int elem   = k & 15;
    long d = ((((long)nTile * kTiles + kTile) * 32) + lane) * 16 + elem;
    dst[d] = f2bf(W[idx]);
}

__global__ void cvt_bf_kernel(const float* __restrict__ src, unsigned short* __restrict__ dst, long n) {
    long i = (long)blockIdx.x * blockDim.x + threadIdx.x;
    if (i < n) dst[i] = f2bf(src[i]);
}

__global__ void zero_state_kernel(float* __restrict__ h, unsigned short* __restrict__ hbf, int n) {
    int i = blockIdx.x * blockDim.x + threadIdx.x;
    if (i < n) { h[i] = 0.0f; hbf[i] = 0; }
}

__global__ void copy_h_kernel(const float* __restrict__ h, float* __restrict__ dst, int n) {
    int i = blockIdx.x * blockDim.x + threadIdx.x;
    if (i < n) dst[i] = h[i];
}

// ---------------------------------------------------------------------------
// Big input-projection GEMM: G[M][N] = A(bf16,row-major,MxK) @ Bp + bias[N]
// Block = 8 waves = 128x64 output tile. The 64-wide B tile (4 KB per k-step)
// is shared by all 8 waves: staged in LDS with double-buffered
// global_load_async_to_lds_b128 (ASYNCcnt), consumed via ds_load.
// ---------------------------------------------------------------------------
__global__ void gemm_gates_kernel(const unsigned short* __restrict__ A,
                                  const unsigned short* __restrict__ Bp,
                                  const float* __restrict__ bias,
                                  float* __restrict__ G,
                                  int M, int N, int K) {
    __shared__ unsigned short smem[2 * 2048];   // 2 buffers x 4 KB (64 cols x 32 k bf16)

    const int lane = threadIdx.x & 31;
    const int wave = threadIdx.x >> 5;
    const int nGroups = N >> 6;
    const int mBlock  = blockIdx.x / nGroups;   // 128-row block
    const int nGrp    = blockIdx.x % nGroups;   // 64-col group
    const int mTile   = mBlock * 8 + wave;      // this wave's 16-row tile
    const int kTiles  = K >> 5;

    // async stage of B tile for k-step kt into buffer bufIdx:
    // thread i copies 16B: chunk j = i/64 (nTile), sub = i%64
    const int jS   = threadIdx.x >> 6;
    const int subS = threadIdx.x & 63;
    const unsigned short* gBase =
        Bp + (((long)(nGrp * 4 + jS) * kTiles) * 512) + subS * 8;
    unsigned ldsBase = (unsigned)(size_t)&smem[jS * 512 + subS * 8];

#define STAGE_B(kt, bufIdx)                                                      \
    do {                                                                         \
        const unsigned short* gp = gBase + (long)(kt) * 512;                     \
        unsigned lp = ldsBase + (unsigned)(bufIdx) * 4096u; /* bytes */          \
        asm volatile("global_load_async_to_lds_b128 %0, %1, off"                 \
                     :: "v"(lp), "v"(gp) : "memory");                            \
    } while (0)

    v8f a0 = {}, a1 = {}, a2 = {}, a3 = {};

    // preload k-step 0
    STAGE_B(0, 0);
    asm volatile("s_wait_asynccnt 0x0" ::: "memory");
    __syncthreads();

    int cur = 0;
    for (int kt = 0; kt < kTiles; ++kt) {
        if (kt + 1 < kTiles) STAGE_B(kt + 1, cur ^ 1);

        v16bf af = load_a_frag(A, K, mTile * 16, kt * 32, lane);
        const unsigned short* sb = &smem[cur * 2048];
        v16bf b0 = __builtin_bit_cast(v16bf, *(const v16us*)&sb[0 * 512 + lane * 16]);
        v16bf b1 = __builtin_bit_cast(v16bf, *(const v16us*)&sb[1 * 512 + lane * 16]);
        v16bf b2 = __builtin_bit_cast(v16bf, *(const v16us*)&sb[2 * 512 + lane * 16]);
        v16bf b3 = __builtin_bit_cast(v16bf, *(const v16us*)&sb[3 * 512 + lane * 16]);
        WMMA_BF16(a0, af, b0);
        WMMA_BF16(a1, af, b1);
        WMMA_BF16(a2, af, b2);
        WMMA_BF16(a3, af, b3);

        // my async issues landed + everyone done reading `cur` before it is
        // overwritten by the next iteration's STAGE_B
        asm volatile("s_wait_asynccnt 0x0" ::: "memory");
        __syncthreads();
        cur ^= 1;
    }
#undef STAGE_B

    int row0 = mTile * 16 + ((lane >> 4) * 8);
    int col0 = nGrp * 64 + (lane & 15);
#pragma unroll
    for (int v = 0; v < 8; ++v) {
        long r = row0 + v;
        G[r * N + col0 +  0] = a0[v] + bias[col0 +  0];
        G[r * N + col0 + 16] = a1[v] + bias[col0 + 16];
        G[r * N + col0 + 32] = a2[v] + bias[col0 + 32];
        G[r * N + col0 + 48] = a3[v] + bias[col0 + 48];
    }
}

// ---------------------------------------------------------------------------
// Scan phase 1: g_hh[:, 0:2H] = h_prev @ w_rz ; r,z = sigmoid(g_ih + g_hh + b)
// stores rhbf = bf16(r * h_prev) and zbuf = z.
// M=64 (batch), N=2048, K=1024 -> 128 waves (16 blocks x 8 waves)
// ---------------------------------------------------------------------------
__global__ void gru_phase1_kernel(const float* __restrict__ h,
                                  const unsigned short* __restrict__ hbf,
                                  const unsigned short* __restrict__ Whhp, // packed, N=3072
                                  const float* __restrict__ Gt,            // G + t*B*3H
                                  const float* __restrict__ b_hh,
                                  unsigned short* __restrict__ rhbf,
                                  float* __restrict__ zbuf) {
    int lane = threadIdx.x & 31;
    int wave = threadIdx.x >> 5;
    int wgid = blockIdx.x * (blockDim.x >> 5) + wave;
    int mTile = wgid >> 5;        // 0..3
    int nGrp  = wgid & 31;        // 0..31  (cols 0..2047)
    const int kTiles = HID >> 5;  // 32

    v8f a0 = {}, a1 = {}, a2 = {}, a3 = {};
    for (int kt = 0; kt < kTiles; ++kt) {
        v16bf af = load_a_frag(hbf, HID, mTile * 16, kt * 32, lane);
        v16bf b0 = load_b_frag(Whhp, kTiles, nGrp * 4 + 0, kt, lane);
        v16bf b1 = load_b_frag(Whhp, kTiles, nGrp * 4 + 1, kt, lane);
        v16bf b2 = load_b_frag(Whhp, kTiles, nGrp * 4 + 2, kt, lane);
        v16bf b3 = load_b_frag(Whhp, kTiles, nGrp * 4 + 3, kt, lane);
        WMMA_BF16(a0, af, b0);
        WMMA_BF16(a1, af, b1);
        WMMA_BF16(a2, af, b2);
        WMMA_BF16(a3, af, b3);
    }
    int row0 = mTile * 16 + ((lane >> 4) * 8);
    int col0 = nGrp * 64 + (lane & 15);
#pragma unroll
    for (int j = 0; j < 4; ++j) {
        v8f acc = (j == 0) ? a0 : (j == 1) ? a1 : (j == 2) ? a2 : a3;
#pragma unroll
        for (int v = 0; v < 8; ++v) {
            int m = row0 + v;
            int n = col0 + j * 16;
            float g = Gt[(long)m * H3 + n] + acc[v] + b_hh[n];
            float s = sigmoidf_(g);
            if (n < HID) {
                rhbf[m * HID + n] = f2bf(s * h[m * HID + n]);
            } else {
                zbuf[m * HID + (n - HID)] = s;
            }
        }
    }
}

// ---------------------------------------------------------------------------
// Scan phase 2: m = (r*h) @ w_o ; h_new = (1-z)*h + z*(o_i + m + b_o)
// M=64, N=1024, K=1024 -> 64 waves (8 blocks x 8 waves)
// ---------------------------------------------------------------------------
__global__ void gru_phase2_kernel(float* __restrict__ h,
                                  unsigned short* __restrict__ hbf,
                                  const unsigned short* __restrict__ rhbf,
                                  const unsigned short* __restrict__ Whhp,
                                  const float* __restrict__ Gt,
                                  const float* __restrict__ b_hh,
                                  const float* __restrict__ zbuf,
                                  float* __restrict__ outf,          // may be null
                                  unsigned short* __restrict__ outbf) { // may be null
    int lane = threadIdx.x & 31;
    int wave = threadIdx.x >> 5;
    int wgid = blockIdx.x * (blockDim.x >> 5) + wave;
    int mTile = wgid >> 4;        // 0..3
    int nGrp  = wgid & 15;        // 0..15 (cols 2048..3071 of packed w_hh)
    const int kTiles = HID >> 5;  // 32

    v8f a0 = {}, a1 = {}, a2 = {}, a3 = {};
    for (int kt = 0; kt < kTiles; ++kt) {
        v16bf af = load_a_frag(rhbf, HID, mTile * 16, kt * 32, lane);
        v16bf b0 = load_b_frag(Whhp, kTiles, 128 + nGrp * 4 + 0, kt, lane);
        v16bf b1 = load_b_frag(Whhp, kTiles, 128 + nGrp * 4 + 1, kt, lane);
        v16bf b2 = load_b_frag(Whhp, kTiles, 128 + nGrp * 4 + 2, kt, lane);
        v16bf b3 = load_b_frag(Whhp, kTiles, 128 + nGrp * 4 + 3, kt, lane);
        WMMA_BF16(a0, af, b0);
        WMMA_BF16(a1, af, b1);
        WMMA_BF16(a2, af, b2);
        WMMA_BF16(a3, af, b3);
    }
    int row0 = mTile * 16 + ((lane >> 4) * 8);
    int col0 = nGrp * 64 + (lane & 15);
#pragma unroll
    for (int j = 0; j < 4; ++j) {
        v8f acc = (j == 0) ? a0 : (j == 1) ? a1 : (j == 2) ? a2 : a3;
#pragma unroll
        for (int v = 0; v < 8; ++v) {
            int m = row0 + v;
            int n = col0 + j * 16;
            float oi = Gt[(long)m * H3 + 2048 + n];
            float ht = oi + acc[v] + b_hh[2048 + n];
            float z  = zbuf[m * HID + n];
            float hp = h[m * HID + n];
            float hn = (1.0f - z) * hp + z * ht;
            h[m * HID + n]   = hn;
            hbf[m * HID + n] = f2bf(hn);
            if (outf)  outf[m * HID + n]  = hn;
            if (outbf) outbf[m * HID + n] = f2bf(hn);
        }
    }
}

// ---------------------------------------------------------------------------
// Host launch
// ---------------------------------------------------------------------------
extern "C" void kernel_launch(void* const* d_in, const int* in_sizes, int n_in,
                              void* d_out, int out_size, void* d_ws, size_t ws_size,
                              hipStream_t stream) {
    const float* x      = (const float*)d_in[0];
    const float* w_ih_0 = (const float*)d_in[1];
    const float* w_hh_0 = (const float*)d_in[2];
    const float* b_ih_0 = (const float*)d_in[3];
    const float* b_hh_0 = (const float*)d_in[4];
    const float* w_ih_1 = (const float*)d_in[5];
    const float* w_hh_1 = (const float*)d_in[6];
    const float* b_ih_1 = (const float*)d_in[7];
    const float* b_hh_1 = (const float*)d_in[8];
    float* out = (float*)d_out;

    // workspace carve-up (256B aligned)
    char* ws = (char*)d_ws;
    size_t off = 0;
    auto carve = [&](size_t bytes) { char* p = ws + off; off = (off + bytes + 255) & ~(size_t)255; return p; };
    const size_t wBytes  = (size_t)DIM * H3 * 2;                  // 6 MB each
    unsigned short* Wih0p  = (unsigned short*)carve(wBytes);
    unsigned short* Whh0p  = (unsigned short*)carve(wBytes);
    unsigned short* Wih1p  = (unsigned short*)carve(wBytes);
    unsigned short* Whh1p  = (unsigned short*)carve(wBytes);
    unsigned short* xbf    = (unsigned short*)carve((size_t)T_STEPS * BATCH * DIM * 2);   // 64 MB
    unsigned short* out0bf = (unsigned short*)carve((size_t)T_STEPS * BATCH * HID * 2);   // 64 MB
    float*          G      = (float*)carve((size_t)T_STEPS * BATCH * H3 * 4);             // 403 MB
    float*          hbuf   = (float*)carve((size_t)BATCH * HID * 4);
    unsigned short* hbf    = (unsigned short*)carve((size_t)BATCH * HID * 2);
    unsigned short* rhbf   = (unsigned short*)carve((size_t)BATCH * HID * 2);
    float*          zbuf   = (float*)carve((size_t)BATCH * HID * 4);
    (void)ws_size; (void)in_sizes; (void)n_in; (void)out_size;

    const int TB = T_STEPS * BATCH;                 // 32768
    const long xElems = (long)TB * DIM;             // 33.5M
    const long wElems = (long)DIM * H3;             // 3.1M

    // --- setup: pack weights, convert x ---
    {
        int blk = 256;
        int g = (int)((wElems + blk - 1) / blk);
        pack_w_kernel<<<g, blk, 0, stream>>>(w_ih_0, Wih0p, DIM, H3);
        pack_w_kernel<<<g, blk, 0, stream>>>(w_hh_0, Whh0p, HID, H3);
        pack_w_kernel<<<g, blk, 0, stream>>>(w_ih_1, Wih1p, HID, H3);
        pack_w_kernel<<<g, blk, 0, stream>>>(w_hh_1, Whh1p, HID, H3);
        cvt_bf_kernel<<<(int)((xElems + blk - 1) / blk), blk, 0, stream>>>(x, xbf, xElems);
    }

    const int gatesBlocks = (TB / 128) * (H3 / 64);      // 256 * 48 = 12288 blocks
    const int stateN = BATCH * HID;                      // 65536

    for (int layer = 0; layer < 2; ++layer) {
        const unsigned short* Wihp = layer ? Wih1p : Wih0p;
        const unsigned short* Whhp = layer ? Whh1p : Whh0p;
        const float* b_ih = layer ? b_ih_1 : b_ih_0;
        const float* b_hh = layer ? b_hh_1 : b_hh_0;
        const unsigned short* Abf = layer ? out0bf : xbf;

        // gates_ih = A @ w_ih + b_ih   (bf16 WMMA, fp32 accumulate)
        gemm_gates_kernel<<<gatesBlocks, 256, 0, stream>>>(Abf, Wihp, b_ih, G, TB, H3, DIM);

        zero_state_kernel<<<stateN / 256, 256, 0, stream>>>(hbuf, hbf, stateN);

        for (int t = 0; t < T_STEPS; ++t) {
            const float* Gt = G + (size_t)t * BATCH * H3;
            gru_phase1_kernel<<<16, 256, 0, stream>>>(hbuf, hbf, Whhp, Gt, b_hh, rhbf, zbuf);
            float* outf = layer ? (out + (size_t)t * BATCH * HID) : nullptr;
            unsigned short* obf = layer ? nullptr : (out0bf + (size_t)t * BATCH * HID);
            gru_phase2_kernel<<<8, 256, 0, stream>>>(hbuf, hbf, rhbf, Whhp, Gt, b_hh, zbuf, outf, obf);
        }

        // h_last for this layer -> tail of d_out: [T*B*H + layer*B*H, ...)
        copy_h_kernel<<<stateN / 256, 256, 0, stream>>>(
            hbuf, out + (size_t)T_STEPS * BATCH * HID + (size_t)layer * BATCH * HID, stateN);
    }
}
